// AttentionLayer_47321949667345
// MI455X (gfx1250) — matrix operand order repaired and checked by reading, not compiled, not run on hardware
//
#include <hip/hip_runtime.h>
#include <math.h>
#include <stdint.h>

// ---------- CDNA5 WMMA / TDM types ----------
typedef __attribute__((ext_vector_type(16))) __bf16 v16bf;
typedef __attribute__((ext_vector_type(8)))  float  v8f;
typedef __attribute__((ext_vector_type(4)))  unsigned int v4u;
typedef __attribute__((ext_vector_type(8)))  int v8i;
typedef __attribute__((ext_vector_type(4)))  int v4i;

union BF16Frag { uint4 q[2]; v16bf v; };

__device__ __forceinline__ unsigned short f2bf(float f) {
    unsigned int u = __float_as_uint(f);
    unsigned int r = 0x7FFFu + ((u >> 16) & 1u);   // round-to-nearest-even
    return (unsigned short)((u + r) >> 16);
}

// ---------- Tensor Data Mover: 2D tile (rows x 32 bf16) global -> LDS ----------
// D# packing per CDNA5 ISA ch.8: group0 = count/lds_addr/global_addr/type,
// group1 = data_size, tensor dims, tile dims, dim0 stride. 2D only (groups 2/3 zero).
// clang-23 toolchain: 6-arg builtin (v4u, v8i, v4i, v4i, v8i, i32 cpol).
__device__ __forceinline__ void tdm_load_2d(unsigned ldsAddr, const void* gptr,
                                            unsigned tensorD0, unsigned tensorD1,
                                            unsigned tileD0, unsigned tileD1,
                                            unsigned long long strideElems) {
    unsigned long long ga = (unsigned long long)(uintptr_t)gptr;
    v4u g0;
    g0[0] = 1u;                                          // count=1, user descriptor
    g0[1] = ldsAddr;                                     // lds_addr (bytes)
    g0[2] = (unsigned)(ga & 0xFFFFFFFFull);              // global_addr[31:0]
    g0[3] = (unsigned)((ga >> 32) & 0x01FFFFFFull) | 0x80000000u;  // addr[56:32] | type=2
    v8i g1;
    g1[0] = (int)(1u << 16);                             // data_size=1 -> 2-byte elems
    g1[1] = (int)((tensorD0 & 0xFFFFu) << 16);           // tensor_dim0[15:0] @bits63:48
    g1[2] = (int)(((tensorD0 >> 16) & 0xFFFFu) | ((tensorD1 & 0xFFFFu) << 16));
    g1[3] = (int)(((tensorD1 >> 16) & 0xFFFFu) | ((tileD0 & 0xFFFFu) << 16)); // tile_dim0
    g1[4] = (int)(tileD1 & 0xFFFFu);                     // tile_dim1 (tile_dim2=0)
    g1[5] = (int)(strideElems & 0xFFFFFFFFull);          // tensor_dim0_stride[31:0]
    g1[6] = (int)((strideElems >> 32) & 0xFFFFull);      // stride[47:32]; dim1_stride lo=0
    g1[7] = 0;
    v4i gz4 = {0, 0, 0, 0};
    v8i gz8 = {0, 0, 0, 0, 0, 0, 0, 0};
    __builtin_amdgcn_tensor_load_to_lds(g0, g1, gz4, gz4, gz8, 0);
}

// ---------- weight-norm + bf16 convert: W[r,:] = g[r]*v[r,:]/||v[r,:]|| ----------
__global__ __launch_bounds__(256)
void wn_kernel(const float* __restrict__ v, const float* __restrict__ g,
               int cols, unsigned short* __restrict__ outB) {
    int r = blockIdx.x;
    const float* row = v + (long long)r * cols;
    float s = 0.f;
    for (int i = threadIdx.x; i < cols; i += 256) { float x = row[i]; s = fmaf(x, x, s); }
    #pragma unroll
    for (int o = 16; o > 0; o >>= 1) s += __shfl_xor(s, o, 32);
    __shared__ float sw[8];
    int lane = threadIdx.x & 31, wid = threadIdx.x >> 5;
    if (lane == 0) sw[wid] = s;
    __syncthreads();
    if (threadIdx.x == 0) { float t = 0.f; for (int i = 0; i < 8; ++i) t += sw[i]; sw[0] = t; }
    __syncthreads();
    float scale = g[r] * rsqrtf(sw[0]);
    unsigned short* orow = outB + (long long)r * cols;
    for (int i = threadIdx.x; i < cols; i += 256) orow[i] = f2bf(row[i] * scale);
}

// ---------- fp32 -> bf16 elementwise ----------
__global__ __launch_bounds__(256)
void cvt_bf16(const float* __restrict__ in, unsigned short* __restrict__ out, long long n4) {
    long long i = (long long)blockIdx.x * 256 + threadIdx.x;
    long long stride = (long long)gridDim.x * 256;
    for (; i < n4; i += stride) {
        float4 v = ((const float4*)in)[i];
        long long o = i * 4;
        out[o] = f2bf(v.x); out[o+1] = f2bf(v.y); out[o+2] = f2bf(v.z); out[o+3] = f2bf(v.w);
    }
}

// ---------- batched transpose + convert: out[c][r] = bf16(in[r][c]) ----------
__global__ __launch_bounds__(256)
void transpose_cvt(const float* __restrict__ in, unsigned short* __restrict__ out,
                   int R, int C, long long inBatch, long long outBatch) {
    __shared__ float tile[32][33];
    long long b = blockIdx.z;
    int r0 = blockIdx.y * 32, c0 = blockIdx.x * 32;
    const float* src = in + b * inBatch;
    unsigned short* dst = out + b * outBatch;
    #pragma unroll
    for (int j = 0; j < 4; ++j) {
        int r = r0 + threadIdx.y + j * 8;
        tile[threadIdx.y + j * 8][threadIdx.x] = src[(long long)r * C + c0 + threadIdx.x];
    }
    __syncthreads();
    #pragma unroll
    for (int j = 0; j < 4; ++j) {
        int c = c0 + threadIdx.y + j * 8;
        dst[(long long)c * R + r0 + threadIdx.x] = f2bf(tile[threadIdx.x][threadIdx.y + j * 8]);
    }
}

// ---------- row softmax over 1024 cols; fp32 in place + bf16 copy ----------
__global__ __launch_bounds__(256)
void softmax_rows(float* __restrict__ attn, unsigned short* __restrict__ attnB) {
    long long row = blockIdx.x;
    float* p = attn + row * 1024;
    unsigned short* pb = attnB + row * 1024;
    float4 x = ((float4*)p)[threadIdx.x];
    int lane = threadIdx.x & 31, wid = threadIdx.x >> 5;
    __shared__ float sm[8], ss[8];

    float m = fmaxf(fmaxf(x.x, x.y), fmaxf(x.z, x.w));
    #pragma unroll
    for (int o = 16; o > 0; o >>= 1) m = fmaxf(m, __shfl_xor(m, o, 32));
    if (lane == 0) sm[wid] = m;
    __syncthreads();
    if (threadIdx.x == 0) { float t = sm[0]; for (int i = 1; i < 8; ++i) t = fmaxf(t, sm[i]); sm[0] = t; }
    __syncthreads();
    m = sm[0];

    float e0 = __expf(x.x - m), e1 = __expf(x.y - m), e2 = __expf(x.z - m), e3 = __expf(x.w - m);
    float s = e0 + e1 + e2 + e3;
    #pragma unroll
    for (int o = 16; o > 0; o >>= 1) s += __shfl_xor(s, o, 32);
    if (lane == 0) ss[wid] = s;
    __syncthreads();
    if (threadIdx.x == 0) { float t = 0.f; for (int i = 0; i < 8; ++i) t += ss[i]; ss[0] = t; }
    __syncthreads();
    float inv = 1.0f / ss[0];

    e0 *= inv; e1 *= inv; e2 *= inv; e3 *= inv;
    ((float4*)p)[threadIdx.x] = make_float4(e0, e1, e2, e3);
    int i4 = threadIdx.x * 4;
    pb[i4] = f2bf(e0); pb[i4+1] = f2bf(e1); pb[i4+2] = f2bf(e2); pb[i4+3] = f2bf(e3);
}

// ---------- TDM + LDS double-buffered WMMA bf16 GEMM ----------
// Block tile 64(M) x 128(N); 8 waves of 32x32. K stepped by 32.
// Wave 0 issues tensor_load_to_lds for the next A(64x32)/B(128x32) panels while
// all waves run v_wmma_f32_16x16x32_bf16 on the current panels from LDS.
// D = A[M,K] * B^T (B stored row-major [N,K]).
// MODE 0: outB = bf16((acc + bias[col] + addm[row,col]) * scale)   (in-proj -> h)
// MODE 1: outF = acc                                               (scores)
// MODE 2: outB = bf16(acc * scale)                                 (ctx)
// MODE 3: outF = (acc + bias[col] + addm[row,col]) * scale         (out-proj)
template <int MODE>
__global__ __launch_bounds__(256)
void wmma_gemm(const unsigned short* __restrict__ A, long long aBatch,
               const unsigned short* __restrict__ BT, long long bBatch,
               int M, int N, int K,
               const float* __restrict__ bias,
               const float* __restrict__ addm, long long addBatch,
               float scale,
               float* __restrict__ outF, unsigned short* __restrict__ outB,
               long long cBatch) {
    __shared__ unsigned short ldsA[2][64 * 32];    // 2 x 4KB
    __shared__ unsigned short ldsB[2][128 * 32];   // 2 x 8KB

    const int tilesN = N >> 7;                 // 128-wide block tiles along N
    const int bm = blockIdx.x / tilesN;
    const int bn = blockIdx.x - bm * tilesN;
    const long long b = blockIdx.y;

    const int wid  = threadIdx.x >> 5;         // 8 waves
    const int wm   = wid >> 2;                 // 0..1 -> M sub-tile
    const int wn   = wid & 3;                  // 0..3 -> N sub-tile
    const int lane = threadIdx.x & 31;
    const int half = lane >> 4, li = lane & 15;

    const unsigned short* Ablk = A + b * aBatch + (long long)(bm * 64) * K;
    const unsigned short* Bblk = BT + b * bBatch + (long long)(bn * 128) * K;
    const int nK = K >> 5;

    if (wid == 0) {
        tdm_load_2d((unsigned)(uintptr_t)&ldsA[0][0], Ablk,
                    (unsigned)K, 64u, 32u, 64u, (unsigned long long)K);
        tdm_load_2d((unsigned)(uintptr_t)&ldsB[0][0], Bblk,
                    (unsigned)K, 128u, 32u, 128u, (unsigned long long)K);
        __builtin_amdgcn_s_wait_tensorcnt(0);
    }
    __syncthreads();

    v8f c00 = {}, c01 = {}, c10 = {}, c11 = {};
    const int arow0 = (wm * 32 + li) * 32;     // element offsets within a panel
    const int arow1 = arow0 + 16 * 32;
    const int brow0 = (wn * 32 + li) * 32;
    const int brow1 = brow0 + 16 * 32;
    const int aoff = half * 8;                 // ISA 16-bit A frag K layout
    const int boff = half * 8;                 // B frag: same ISA K layout as A

    for (int i = 0; i < nK; ++i) {
        const unsigned short* pa = &ldsA[i & 1][0];
        const unsigned short* pb = &ldsB[i & 1][0];
        if (wid == 0 && (i + 1) < nK) {
            int nxt = (i + 1) & 1;
            tdm_load_2d((unsigned)(uintptr_t)&ldsA[nxt][0], Ablk + (long long)(i + 1) * 32,
                        (unsigned)K, 64u, 32u, 64u, (unsigned long long)K);
            tdm_load_2d((unsigned)(uintptr_t)&ldsB[nxt][0], Bblk + (long long)(i + 1) * 32,
                        (unsigned)K, 128u, 32u, 128u, (unsigned long long)K);
        }
        BF16Frag fa0, fa1, fb0, fb1;
        fa0.q[0] = *(const uint4*)(pa + arow0 + aoff);
        fa0.q[1] = *(const uint4*)(pa + arow0 + 16 + aoff);
        fa1.q[0] = *(const uint4*)(pa + arow1 + aoff);
        fa1.q[1] = *(const uint4*)(pa + arow1 + 16 + aoff);
        fb0.q[0] = *(const uint4*)(pb + brow0 + boff);
        fb0.q[1] = *(const uint4*)(pb + brow0 + boff + 16);
        fb1.q[0] = *(const uint4*)(pb + brow1 + boff);
        fb1.q[1] = *(const uint4*)(pb + brow1 + boff + 16);
        c00 = __builtin_amdgcn_wmma_f32_16x16x32_bf16(false, fa0.v, false, fb0.v, (short)0, c00, false, false);
        c01 = __builtin_amdgcn_wmma_f32_16x16x32_bf16(false, fa0.v, false, fb1.v, (short)0, c01, false, false);
        c10 = __builtin_amdgcn_wmma_f32_16x16x32_bf16(false, fa1.v, false, fb0.v, (short)0, c10, false, false);
        c11 = __builtin_amdgcn_wmma_f32_16x16x32_bf16(false, fa1.v, false, fb1.v, (short)0, c11, false, false);
        if (wid == 0) __builtin_amdgcn_s_wait_tensorcnt(0);  // next panels landed
        __syncthreads();                                     // safe to swap buffers
    }

    float* of = outF ? outF + b * cBatch : nullptr;
    unsigned short* ob = outB ? outB + b * cBatch : nullptr;
    const float* am = addm ? addm + b * addBatch : nullptr;

    v8f accs[2][2] = { { c00, c01 }, { c10, c11 } };
    #pragma unroll
    for (int ti = 0; ti < 2; ++ti) {
        #pragma unroll
        for (int tj = 0; tj < 2; ++tj) {
            int col = bn * 128 + wn * 32 + tj * 16 + li;
            #pragma unroll
            for (int e = 0; e < 8; ++e) {
                int row = bm * 64 + wm * 32 + ti * 16 + half * 8 + e;  // ISA C/D frag layout
                long long off = (long long)row * N + col;
                float val = accs[ti][tj][e];
                if (MODE == 0)      { val = (val + bias[col] + am[off]) * scale; ob[off] = f2bf(val); }
                else if (MODE == 1) { of[off] = val; }
                else if (MODE == 2) { ob[off] = f2bf(val * scale); }
                else                { val = (val + bias[col] + am[off]) * scale; of[off] = val; }
            }
        }
    }
}

extern "C" void kernel_launch(void* const* d_in, const int* in_sizes, int n_in,
                              void* d_out, int out_size, void* d_ws, size_t ws_size,
                              hipStream_t stream) {
    const float* x   = (const float*)d_in[0];
    const float* te  = (const float*)d_in[1];
    const float* ek  = (const float*)d_in[2];   // [N,E,S]
    const float* ev  = (const float*)d_in[3];   // [N,S,E]
    const float* ipv = (const float*)d_in[4];
    const float* ipg = (const float*)d_in[5];
    const float* ipb = (const float*)d_in[6];
    const float* opv = (const float*)d_in[7];
    const float* opg = (const float*)d_in[8];
    const float* opb = (const float*)d_in[9];

    const long long NB = 16, T = 1024, C = 1024, E = 1024, S = 1024;
    const long long MAT = T * C;                 // 1M elems per batch matrix
    float* outp = (float*)d_out;                 // [16,1024,1024] fp32
    float* attn = outp + NB * MAT;               // [16,1024,1024] fp32

    // workspace: regA/regB are reused (serial stream ordering makes this safe)
    char* w = (char*)d_ws;
    const long long REG = NB * MAT * 2;          // 32MB
    unsigned short* regA  = (unsigned short*)(w);            // x_bf, later attn_bf
    unsigned short* regB  = (unsigned short*)(w + REG);      // h_bf, later ctx_bf
    unsigned short* keysT = (unsigned short*)(w + 2 * REG);  // [N,S,E]
    unsigned short* valsT = (unsigned short*)(w + 3 * REG);  // [N,E,S]
    unsigned short* WiB   = (unsigned short*)(w + 4 * REG);             // [E,C]
    unsigned short* WoB   = (unsigned short*)(w + 4 * REG + E * C * 2); // [C,E]

    const float sh = 0.70710678118654752f;       // sqrt(0.5)
    const float ctxScale = 32.0f;                // s*sqrt(1/s), s=1024

    wn_kernel<<<(int)E, 256, 0, stream>>>(ipv, ipg, (int)C, WiB);
    wn_kernel<<<(int)C, 256, 0, stream>>>(opv, opg, (int)E, WoB);
    cvt_bf16<<<8192, 256, 0, stream>>>(x, regA, NB * MAT / 4);
    dim3 tb(32, 8), tg(32, 32, (unsigned)NB);
    transpose_cvt<<<tg, tb, 0, stream>>>(ek, keysT, (int)E, (int)S, MAT, MAT); // -> keysT[S,E]
    transpose_cvt<<<tg, tb, 0, stream>>>(ev, valsT, (int)S, (int)E, MAT, MAT); // -> valsT[E,S]

    { // h = (x@Wi^T + b + te)*sqrt(.5)  -> bf16 regB   M=16384,N=1024,K=1024
        int M = (int)(NB * T), N_ = (int)E, K = (int)C;
        int blocks = (M / 64) * (N_ / 128);
        wmma_gemm<0><<<dim3(blocks, 1), 256, 0, stream>>>(regA, 0, WiB, 0, M, N_, K,
                                                          ipb, te, 0, sh, nullptr, regB, 0);
    }
    { // scores = h @ keys -> fp32 attn area (batched 16)
        int M = (int)T, N_ = (int)S, K = (int)E;
        int blocks = (M / 64) * (N_ / 128);
        wmma_gemm<1><<<dim3(blocks, (unsigned)NB), 256, 0, stream>>>(regB, MAT, keysT, MAT, M, N_, K,
                                                                     nullptr, nullptr, 0, 1.0f, attn, nullptr, MAT);
    }
    softmax_rows<<<(int)(NB * T), 256, 0, stream>>>(attn, regA);   // attn fp32 in place + bf16 regA
    { // ctx = 32 * attn @ V -> bf16 regB (batched 16)
        int M = (int)T, N_ = (int)E, K = (int)S;
        int blocks = (M / 64) * (N_ / 128);
        wmma_gemm<2><<<dim3(blocks, (unsigned)NB), 256, 0, stream>>>(regA, MAT, valsT, MAT, M, N_, K,
                                                                     nullptr, nullptr, 0, ctxScale, nullptr, regB, MAT);
    }
    { // out = (ctx@Wo^T + b + x)*sqrt(.5) -> fp32 d_out
        int M = (int)(NB * T), N_ = (int)C, K = (int)E;
        int blocks = (M / 64) * (N_ / 128);
        wmma_gemm<3><<<dim3(blocks, 1), 256, 0, stream>>>(regB, 0, WoB, 0, M, N_, K,
                                                          opb, x, 0, sh, outp, nullptr, 0);
    }
    (void)in_sizes; (void)n_in; (void)out_size; (void)ws_size;
}
